// LSTMPredictor_180388627103
// MI455X (gfx1250) — matrix operand order repaired and checked by reading, not compiled, not run on hardware
//
#include <hip/hip_runtime.h>

// Problem sizes (fixed by the reference).
#define LT 8192   // timesteps
#define LI 256    // input dim
#define LH 1024   // hidden dim
#define LG 4096   // 4*H gate dim
#define NWG 64    // persistent workgroups for the scan (each owns LH/NWG=16 hidden units)
#define CH 16     // hidden units per workgroup

typedef __attribute__((ext_vector_type(16))) __bf16 v16bf;
typedef __attribute__((ext_vector_type(8)))  float  v8f;
typedef unsigned int u32;
typedef __attribute__((ext_vector_type(4))) u32 u32x4;
typedef __attribute__((ext_vector_type(4))) int i32x4;
typedef __attribute__((ext_vector_type(8))) int i32x8;

union ABOp { uint4 q[2]; v16bf v; };

__device__ inline v8f wmma_bf16(v16bf a, v16bf b, v8f c) {
  // D = A(16x32 bf16) * B(32x16 bf16) + C(16x16 f32)
  return __builtin_amdgcn_wmma_f32_16x16x32_bf16(false, a, false, b, (short)0, c, false, false);
}

__device__ inline unsigned short f2bf(float f) {  // round-to-nearest-even f32 -> bf16
  unsigned u = __float_as_uint(f);
  u += 0x7FFFu + ((u >> 16) & 1u);
  return (unsigned short)(u >> 16);
}

__device__ inline float sigmoidf(float x) { return 1.0f / (1.0f + expf(-x)); }

// ---------------------------------------------------------------- conversions
__global__ __launch_bounds__(256) void cvt_f32_bf16(const float* __restrict__ src,
                                                    unsigned short* __restrict__ dst, int n) {
  int i = blockIdx.x * 256 + threadIdx.x;
  if (i < n) dst[i] = f2bf(src[i]);
}

__global__ __launch_bounds__(256) void init_k(int* ctrl, unsigned short* hbf) {
  int tid = threadIdx.x;
  if (tid < 64) ctrl[tid] = 0;
  for (int i = tid; i < 2 * LH; i += 256) hbf[i] = 0;
}

// ------------------------------------------------- phase 1: x_gates WMMA GEMM
// xg[t][g] = sum_k seq[t][k] * W_ih[g][k] + (b_ih+b_hh)[g]
// One wave computes one 16(t) x 16(g) tile, K=256 in 8 WMMAs. Waves of a block
// share the same 16 t-rows, so A-loads hit L0.
__global__ __launch_bounds__(256) void xgates_gemm(const unsigned short* __restrict__ seqb, // [LT][LI]
                                                   const unsigned short* __restrict__ wihb, // [LG][LI]
                                                   const float* __restrict__ bih,
                                                   const float* __restrict__ bhh,
                                                   float* __restrict__ xg) {                 // [LT][LG]
  const int lane = threadIdx.x & 31;
  const int wave = threadIdx.x >> 5;
  const int tile = blockIdx.x * 8 + wave;       // (LT/16)*(LG/16) = 131072 tiles
  const int tm = tile >> 8;                     // LG/16 = 256 tiles in N
  const int tn = tile & 255;
  const int t0 = tm * 16, g0 = tn * 16;
  const int m  = lane & 15;
  const bool hi = lane >= 16;

  const unsigned short* arow = seqb + (size_t)(t0 + m) * LI;  // A: lane = M row
  const unsigned short* brow = wihb + (size_t)(g0 + m) * LI;  // B: lane = N col
  const int akb = hi ? 8 : 0;    // A: K {0-7,16-23} vs {8-15,24-31}
  const int bkb = hi ? 8 : 0;    // B: K {0-7,16-23} vs {8-15,24-31}

  v8f acc = {};
  for (int k0 = 0; k0 < LI; k0 += 32) {
    ABOp A, B;
    A.q[0] = *(const uint4*)(arow + k0 + akb);
    A.q[1] = *(const uint4*)(arow + k0 + 16 + akb);
    B.q[0] = *(const uint4*)(brow + k0 + bkb);
    B.q[1] = *(const uint4*)(brow + k0 + bkb + 16);
    acc = wmma_bf16(A.v, B.v, acc);
  }
  const float bias = bih[g0 + m] + bhh[g0 + m];
  const int trow = t0 + (hi ? 8 : 0);   // C/D: VGPR r, lanes0-15 -> M=r, lanes16-31 -> M=8+r
#pragma unroll
  for (int r = 0; r < 8; ++r)
    xg[(size_t)(trow + r) * LG + g0 + m] = acc[r] + bias;
}

// ---------------------------------------------------- phase 2: persistent scan
// 64 workgroups, each owns 16 hidden units. Its 64 W_hh rows (4 gates x 16)
// live in LDS (128 KB, bf16) for the whole kernel, fetched by a single TDM
// tensor_load_to_lds (3D tile: 1024 x 16 rows x 4 gate blocks). Per step:
// async-load h_{t-1} broadcast into LDS, WMMA GEMV with h replicated across
// the A M-rows, pointwise cell update, publish 16 h values, global barrier.
__global__ __launch_bounds__(256) void lstm_scan(const unsigned short* __restrict__ whhb, // [LG][LH]
                                                 const float* __restrict__ xg,            // [LT][LG]
                                                 float* __restrict__ hs,                  // [LT][LH]
                                                 unsigned short* __restrict__ hbf,        // [2][LH]
                                                 int* __restrict__ ctrl) {
  extern __shared__ char smem[];
  unsigned short* Wlds = (unsigned short*)smem;            // [64][1024] bf16 = 128 KB
  unsigned short* hA   = Wlds + 64 * LH;                   // [1024] bf16 staging of h_{t-1}
  float* partial       = (float*)(hA + LH);                // [4 tiles][2 halves][16]
  float* cst           = partial + 4 * 2 * 16;             // [16] cell state (f32)

  const int wgid  = blockIdx.x;
  const int hbase = wgid * CH;
  const int tid   = threadIdx.x;
  const int lane  = tid & 31;
  const int wave  = tid >> 5;

  // ---- Preload this WG's 64 gate rows of W_hh into LDS.
#if __has_builtin(__builtin_amdgcn_tensor_load_to_lds)
  if (tid < 32) {  // wave 0 issues one TDM op for the whole 128 KB tile set
    const unsigned long long ga =
        (unsigned long long)(uintptr_t)(whhb + (size_t)hbase * LH);
    u32x4 g0;
    g0[0] = 1u;                                   // count=1, user descriptor
    g0[1] = (u32)(uintptr_t)Wlds;                 // lds_addr (bytes)
    g0[2] = (u32)ga;                              // global_addr[31:0]
    g0[3] = (u32)(ga >> 32) | (2u << 30);         // global_addr[56:32] | type=2
    i32x8 g1;
    g1[0] = (int)(1u << 16);                      // wg_mask=0, data_size=2B
    g1[1] = (int)(1024u << 16);                   // tensor_dim0 = 1024 (row len)
    g1[2] = (int)(4096u << 16);                   // tensor_dim1 = 4096 (rows)
    g1[3] = (int)(1024u << 16);                   // tile_dim0 = 1024
    g1[4] = (int)(16u | (4u << 16));              // tile_dim1 = 16, tile_dim2 = 4
    g1[5] = 1024;                                 // tensor_dim0_stride = 1024 elems
    g1[6] = 0;                                    // strides mid bits
    g1[7] = 0x10;                                 // tensor_dim1_stride = 0x100000 elems
    i32x4 g2; g2[0] = 4; g2[1] = 0; g2[2] = 0; g2[3] = 0;   // tensor_dim2 = 4
    i32x4 g3; g3[0] = 0; g3[1] = 0; g3[2] = 0; g3[3] = 0;
    i32x8 g4 = {0, 0, 0, 0, 0, 0, 0, 0};                    // unused extra group
    __builtin_amdgcn_tensor_load_to_lds(g0, g1, g2, g3, g4, 0);
  }
  __builtin_amdgcn_s_wait_tensorcnt(0);
#else
  for (int idx = tid * 8; idx < 64 * LH; idx += 256 * 8) {
    const int r = idx >> 10, k = idx & (LH - 1);
    const int gsel = r >> 4, n = r & 15;
    *(uint4*)(Wlds + idx) =
        *(const uint4*)(whhb + (size_t)(gsel * LH + hbase + n) * LH + k);
  }
#endif
  if (tid < CH) cst[tid] = 0.0f;
  __syncthreads();

  const int  tileg = wave & 3;        // which gate block (i,f,g,o)
  const int  half  = wave >> 2;       // K half: [0,512) or [512,1024)
  const int  n     = lane & 15;
  const bool hi    = lane >= 16;
  const unsigned short* brow = Wlds + (size_t)(tileg * 16 + n) * LH + (hi ? 8 : 0);
  const unsigned short* ha0  = hA + (hi ? 8 : 0);
  const int kbeg = half * (LH / 2), kend = kbeg + (LH / 2);

  const unsigned       hA_off   = (unsigned)(uintptr_t)hA;
  const unsigned long long hbfb = (unsigned long long)(uintptr_t)hbf;

  for (int t = 0; t < LT; ++t) {
    const int pr = (t + 1) & 1;   // parity holding h_{t-1}
    const int pw = t & 1;         // parity to publish h_t

    // Prefetch this step's x_gates row (64 f32 across 4 gate blocks).
    if (tid < 4) __builtin_prefetch(xg + (size_t)t * LG + tid * LH + hbase, 0, 0);

    // Stage h_{t-1} (bf16, 2 KB) global -> LDS without touching VGPR data:
    // async copy, ASYNCcnt-tracked. 256 lanes x 8 B.
    {
      const unsigned ldsd = hA_off + tid * 8;
      const unsigned gvo  = (unsigned)(pr * (LH * 2) + tid * 8);
      asm volatile("global_load_async_to_lds_b64 %0, %1, %2"
                   :: "v"(ldsd), "v"(gvo), "s"(hbfb) : "memory");
      asm volatile("s_wait_asynccnt 0x0" ::: "memory");
    }
    __syncthreads();

    // GEMV slice via WMMA: A = replicated h, B = W_hh^T tile from LDS.
    v8f acc = {};
#pragma unroll 4
    for (int k0 = kbeg; k0 < kend; k0 += 32) {
      ABOp A, B;
      A.q[0] = *(const uint4*)(ha0 + k0);
      A.q[1] = *(const uint4*)(ha0 + k0 + 16);
      B.q[0] = *(const uint4*)(brow + k0);
      B.q[1] = *(const uint4*)(brow + k0 + 16);
      acc = wmma_bf16(A.v, B.v, acc);
    }
    if (lane < 16) partial[(tileg * 2 + half) * 16 + n] = acc[0]; // D row M=0
    __syncthreads();

    // Pointwise LSTM cell for this WG's 16 hidden units.
    if (tid < CH) {
      const int j = tid;
      const float* xrow = xg + (size_t)t * LG;
      const float di = partial[(0 * 2 + 0) * 16 + j] + partial[(0 * 2 + 1) * 16 + j];
      const float df = partial[(1 * 2 + 0) * 16 + j] + partial[(1 * 2 + 1) * 16 + j];
      const float dg = partial[(2 * 2 + 0) * 16 + j] + partial[(2 * 2 + 1) * 16 + j];
      const float dz = partial[(3 * 2 + 0) * 16 + j] + partial[(3 * 2 + 1) * 16 + j];
      const float gi = sigmoidf(xrow[0 * LH + hbase + j] + di);
      const float gf = sigmoidf(xrow[1 * LH + hbase + j] + df);
      const float gg = tanhf   (xrow[2 * LH + hbase + j] + dg);
      const float go = sigmoidf(xrow[3 * LH + hbase + j] + dz);
      const float c  = gf * cst[j] + gi * gg;
      cst[j] = c;
      const float h = go * tanhf(c);
      hs[(size_t)t * LH + hbase + j] = h;
      hbf[pw * LH + hbase + j] = f2bf(h);
    }

    // Grid-wide barrier: monotonic counter (no reset race), agent scope.
    __syncthreads();
    if (tid == 0) {
      __threadfence();
      __hip_atomic_fetch_add(&ctrl[0], 1, __ATOMIC_ACQ_REL, __HIP_MEMORY_SCOPE_AGENT);
      const int target = (t + 1) * NWG;
      while (__hip_atomic_load(&ctrl[0], __ATOMIC_ACQUIRE, __HIP_MEMORY_SCOPE_AGENT) < target)
        __builtin_amdgcn_s_sleep(2);
    }
    __syncthreads();
  }
}

// -------------------------------------------------- phase 3: output projection
__global__ __launch_bounds__(256) void out_proj(const float* __restrict__ hs,
                                                const float* __restrict__ wout,
                                                const float* __restrict__ bout,
                                                float* __restrict__ out) {
  const int t = blockIdx.x * 8 + (threadIdx.x >> 5);
  const int lane = threadIdx.x & 31;
  const float* hr = hs + (size_t)t * LH;
  float s = 0.0f;
  for (int k = lane; k < LH; k += 32) s += hr[k] * wout[k];
#pragma unroll
  for (int off = 16; off > 0; off >>= 1) s += __shfl_xor(s, off, 32);
  if (lane == 0) out[t] = sigmoidf(s + bout[0]);
}

// ----------------------------------------------------------------- launcher
extern "C" void kernel_launch(void* const* d_in, const int* in_sizes, int n_in,
                              void* d_out, int out_size, void* d_ws, size_t ws_size,
                              hipStream_t stream) {
  const float* seq  = (const float*)d_in[0];
  const float* wih  = (const float*)d_in[1];
  const float* whh  = (const float*)d_in[2];
  const float* bih  = (const float*)d_in[3];
  const float* bhh  = (const float*)d_in[4];
  const float* wout = (const float*)d_in[5];
  const float* bout = (const float*)d_in[6];
  float* out = (float*)d_out;

  char* ws = (char*)d_ws;
  size_t cur = 0;
  auto alloc = [&](size_t bytes) -> void* {
    void* p = ws + cur;
    cur += (bytes + 255) & ~(size_t)255;
    return p;
  };
  int*            ctrl = (int*)alloc(256);
  unsigned short* hbf  = (unsigned short*)alloc((size_t)2 * LH * 2);
  unsigned short* seqb = (unsigned short*)alloc((size_t)LT * LI * 2);
  unsigned short* wihb = (unsigned short*)alloc((size_t)LG * LI * 2);
  unsigned short* whhb = (unsigned short*)alloc((size_t)LG * LH * 2);
  float*          xg   = (float*)alloc((size_t)LT * LG * 4);
  float*          hs   = (float*)alloc((size_t)LT * LH * 4);

  cvt_f32_bf16<<<(LT * LI + 255) / 256, 256, 0, stream>>>(seq, seqb, LT * LI);
  cvt_f32_bf16<<<(LG * LI + 255) / 256, 256, 0, stream>>>(wih, wihb, LG * LI);
  cvt_f32_bf16<<<(LG * LH + 255) / 256, 256, 0, stream>>>(whh, whhb, LG * LH);
  init_k<<<1, 256, 0, stream>>>(ctrl, hbf);

  xgates_gemm<<<(LT / 16) * (LG / 16) / 8, 256, 0, stream>>>(seqb, wihb, bih, bhh, xg);

  const size_t smem = (size_t)64 * LH * 2   // Wlds 128 KB
                    + (size_t)LH * 2        // hA
                    + 4 * 2 * 16 * 4        // partials
                    + CH * 4;               // cell state
  (void)hipFuncSetAttribute((const void*)lstm_scan,
                            hipFuncAttributeMaxDynamicSharedMemorySize, (int)smem);
  lstm_scan<<<NWG, 256, smem, stream>>>(whhb, xg, hs, hbf, ctrl);

  out_proj<<<LT / 8, 256, 0, stream>>>(hs, wout, bout, out);
}
